// _UniMatchTransformerLayer_28509992911250
// MI455X (gfx1250) — matrix operand order repaired and checked by reading, not compiled, not run on hardware
//
#include <hip/hip_runtime.h>
#include <hip/hip_bf16.h>
#include <math.h>

// ---------------------------------------------------------------------------
// UniMatch transformer layer for gfx1250 (MI455X), all GEMMs on
// v_wmma_f32_16x16x32_bf16 (fp32 accumulate), flash-attention windowed path.
// ---------------------------------------------------------------------------

typedef __attribute__((ext_vector_type(16))) __bf16 v16bf;
typedef __attribute__((ext_vector_type(8)))  __bf16 v8bf;
typedef __attribute__((ext_vector_type(8)))  float  v8f;

#define D_      128
#define HGT_    64
#define WID_    128
#define HH_     32
#define WW_     64
#define LW_     2048            // tokens per window
#define NTOK_   32768
#define HID_    1024
#define LN_EPS_ 1e-5f
#define SCALE_  0.08838834764831845f   // 1/sqrt(128)
#define PLDS_STRIDE_ 40                 // 16x32 bf16 tile, padded rows

static __device__ __forceinline__ __bf16 f2bf(float f) { return (__bf16)f; }

// A-matrix fragment (16x32, bf16) from an fp32 row-major row.
// lane: m = lane&15, half = lane>>4.  e<8 -> k=8*half+e ; e>=8 -> k=16+8*half+(e-8)
static __device__ __forceinline__ v16bf a_frag_f32(const float* rowp, int kbase, int half) {
  const float* p0 = rowp + kbase + 8 * half;
  const float* p1 = rowp + kbase + 16 + 8 * half;
  v16bf a;
#pragma unroll
  for (int i = 0; i < 8; ++i) { a[i] = f2bf(p0[i]); a[8 + i] = f2bf(p1[i]); }
  return a;
}

// A-matrix fragment from a bf16 row-major row (two contiguous 16B chunks).
static __device__ __forceinline__ v16bf a_frag_bf(const __bf16* rowp, int kbase, int half) {
  v8bf lo = *(const v8bf*)(rowp + kbase + 8 * half);
  v8bf hi = *(const v8bf*)(rowp + kbase + 16 + 8 * half);
  v16bf a;
#pragma unroll
  for (int i = 0; i < 8; ++i) { a[i] = lo[i]; a[8 + i] = hi[i]; }
  return a;
}

// B-matrix fragment (32x16, bf16): lane n holds e<8 -> k=kbase+8*half+e ; e>=8 -> k=kbase+16+8*half+(e-8).
// rowp must point at the contiguous K-major row for output column n.
static __device__ __forceinline__ v16bf b_frag(const __bf16* rowp, int kbase, int half) {
  v8bf lo = *(const v8bf*)(rowp + kbase + 8 * half);
  v8bf hi = *(const v8bf*)(rowp + kbase + 16 + 8 * half);
  v16bf b;
#pragma unroll
  for (int i = 0; i < 8; ++i) { b[i] = lo[i]; b[8 + i] = hi[i]; }
  return b;
}

static __device__ __forceinline__ v8f wmma_bf16(v16bf a, v16bf b, v8f c) {
  return __builtin_amdgcn_wmma_f32_16x16x32_bf16(false, a, false, b, (short)0, c,
                                                 false, false);
}

// window (win,l) -> flat token index, including the shifted-window roll.
static __device__ __forceinline__ int map_token(int win, int l, int shift) {
  int b = win >> 2, wy = (win >> 1) & 1, wx = win & 1;
  int iy = l >> 6, ix = l & 63;          // ww = 64
  int y = wy * HH_ + iy, x = wx * WW_ + ix;
  if (shift) { y = (y + HH_ / 2) & (HGT_ - 1); x = (x + WW_ / 2) & (WID_ - 1); }
  return (b << 13) | (y << 7) | x;       // b*8192 + y*128 + x
}

// ---------------------------------------------------------------------------
// K0: convert + transpose all weights to bf16 so WMMA B-fragments are
// contiguous 16B loads.  Wt[dout][din].
// ---------------------------------------------------------------------------
__global__ void k_prep_weights(const float* __restrict__ Wq, const float* __restrict__ Wk,
                               const float* __restrict__ Wv, const float* __restrict__ Wm,
                               const float* __restrict__ W1, const float* __restrict__ W2,
                               __bf16* __restrict__ Wqt, __bf16* __restrict__ Wkt,
                               __bf16* __restrict__ Wvt, __bf16* __restrict__ Wmt,
                               __bf16* __restrict__ W1t, __bf16* __restrict__ W2t) {
  int i = blockIdx.x * blockDim.x + threadIdx.x;
  if (i < 4 * 128 * 128) {
    int m = i >> 14, j = i & 16383;
    int dout = j >> 7, din = j & 127;
    const float* s = (m == 0) ? Wq : (m == 1) ? Wk : (m == 2) ? Wv : Wm;
    __bf16* d = (m == 0) ? Wqt : (m == 1) ? Wkt : (m == 2) ? Wvt : Wmt;
    d[dout * 128 + din] = f2bf(s[din * 128 + dout]);
  } else if (i < 65536 + 1024 * 256) {
    int j = i - 65536;
    int h = j >> 8, k = j & 255;                 // W1: (256,1024) -> W1t[h][k]
    W1t[h * 256 + k] = f2bf(W1[k * 1024 + h]);
  } else if (i < 65536 + 262144 + 131072) {
    int j = i - 65536 - 262144;
    int dout = j >> 10, h = j & 1023;            // W2: (1024,128) -> W2t[d][h]
    W2t[dout * 1024 + h] = f2bf(W2[h * 128 + dout]);
  }
}

// ---------------------------------------------------------------------------
// K1: QKV projection directly into windowed layout.
//   Qw, Kw : [win][l][d] row-major bf16
//   Vt     : [win][d][l] transposed bf16   (B-fragment friendly for P*V)
// One wave = one 16-token tile of one matrix (q/k/v).
// ---------------------------------------------------------------------------
__global__ void __launch_bounds__(32)
k_qkv_proj(const float* __restrict__ src, const __bf16* __restrict__ Wqt,
           const __bf16* __restrict__ Wkt, const __bf16* __restrict__ Wvt,
           __bf16* __restrict__ Qw, __bf16* __restrict__ Kw,
           __bf16* __restrict__ Vt, const int* __restrict__ p_shift) {
  const int gid  = blockIdx.x;
  const int tile = gid & 2047;            // 2048 row tiles
  const int mat  = gid >> 11;             // 0=Q 1=K 2=V
  const int win  = tile >> 7;             // 128 tiles per window
  const int l0   = (tile & 127) << 4;
  const int lane = threadIdx.x;
  const int n = lane & 15, half = lane >> 4;
  const int shift = __builtin_amdgcn_readfirstlane(*p_shift);

  // A fragments: gathered (rolled) token rows, fp32 -> bf16.
  const int tok = map_token(win, l0 + n, shift);
  const float* rowp = src + (size_t)tok * D_;
  v16bf a[4];
#pragma unroll
  for (int kc = 0; kc < 4; ++kc) a[kc] = a_frag_f32(rowp, kc * 32, half);

  const __bf16* Wt = (mat == 0) ? Wqt : (mat == 1) ? Wkt : Wvt;

#pragma unroll
  for (int nt = 0; nt < 8; ++nt) {
    v8f c = {};
    const __bf16* wrow = Wt + (size_t)(nt * 16 + n) * D_;
#pragma unroll
    for (int kc = 0; kc < 4; ++kc)
      c = wmma_bf16(a[kc], b_frag(wrow, kc * 32, half), c);
    const int d = nt * 16 + n;
    if (mat == 2) {
      // transposed store: this lane's 8 C values are 8 consecutive tokens
      v8bf vv;
#pragma unroll
      for (int r = 0; r < 8; ++r) vv[r] = f2bf(c[r]);
      *(v8bf*)(Vt + (size_t)(win * D_ + d) * LW_ + l0 + 8 * half) = vv;
    } else {
      __bf16* dst = (mat == 0) ? Qw : Kw;
#pragma unroll
      for (int r = 0; r < 8; ++r)
        dst[((size_t)win * LW_ + l0 + r + 8 * half) * D_ + d] = f2bf(c[r]);
    }
  }
}

// ---------------------------------------------------------------------------
// K2: flash attention per window.  One wave = 16 queries, streams 64 key
// blocks of 32.  Scores -> online softmax -> P relayout via LDS -> P*V.
// ---------------------------------------------------------------------------
__global__ void __launch_bounds__(32)
k_attn(const __bf16* __restrict__ Qw, const __bf16* __restrict__ Kw,
       const __bf16* __restrict__ Vt, const float* __restrict__ mask,
       __bf16* __restrict__ Aout, const int* __restrict__ p_shift) {
  __shared__ __align__(16) __bf16 Plds[16 * PLDS_STRIDE_];
  const int win = blockIdx.x >> 7;
  const int q0  = (blockIdx.x & 127) << 4;
  const int lane = threadIdx.x;
  const int n = lane & 15, half = lane >> 4;
  const int shift = __builtin_amdgcn_readfirstlane(*p_shift);

  const __bf16* qrow = Qw + ((size_t)win * LW_ + q0 + n) * D_;
  v16bf qa[4];
#pragma unroll
  for (int kc = 0; kc < 4; ++kc) qa[kc] = a_frag_bf(qrow, kc * 32, half);

  const float* maskp = mask + (size_t)(win & 3) * LW_ * LW_;

  float m_i[8], l_i[8];
  v8f o[8];
  v8f z = {};
#pragma unroll
  for (int r = 0; r < 8; ++r) { m_i[r] = -3.0e38f; l_i[r] = 0.f; }
#pragma unroll
  for (int nt = 0; nt < 8; ++nt) o[nt] = z;

  for (int j = 0; j < 64; ++j) {
    const int k0 = j * 32;
    // prefetch next key block rows into cache
    if (j + 1 < 64) {
      __builtin_prefetch(Kw + ((size_t)win * LW_ + k0 + 32 + n) * D_, 0, 0);
      __builtin_prefetch(Vt + (size_t)(win * D_ + n) * LW_ + k0 + 32, 0, 0);
    }
    float p0v[8], p1v[8], tmax[8], tsum[8];
#pragma unroll
    for (int st = 0; st < 2; ++st) {
      v8f s = z;
      const __bf16* krow = Kw + ((size_t)win * LW_ + k0 + st * 16 + n) * D_;
#pragma unroll
      for (int kc = 0; kc < 4; ++kc)
        s = wmma_bf16(qa[kc], b_frag(krow, kc * 32, half), s);
      float* pv = st ? p1v : p0v;
#pragma unroll
      for (int r = 0; r < 8; ++r) {
        float sv = s[r] * SCALE_;
        if (shift) {
          int qi = q0 + r + 8 * half;
          int ki = k0 + st * 16 + n;
          sv += maskp[(size_t)qi * LW_ + ki];
        }
        pv[r] = sv;
      }
    }
    // per-row max across the 16-lane half-groups
#pragma unroll
    for (int r = 0; r < 8; ++r) tmax[r] = fmaxf(p0v[r], p1v[r]);
#pragma unroll
    for (int off = 1; off < 16; off <<= 1) {
#pragma unroll
      for (int r = 0; r < 8; ++r) tmax[r] = fmaxf(tmax[r], __shfl_xor(tmax[r], off));
    }
    float alpha[8];
#pragma unroll
    for (int r = 0; r < 8; ++r) {
      float mn = fmaxf(m_i[r], tmax[r]);
      alpha[r] = __expf(m_i[r] - mn);
      m_i[r] = mn;
      p0v[r] = __expf(p0v[r] - mn);
      p1v[r] = __expf(p1v[r] - mn);
      tsum[r] = p0v[r] + p1v[r];
    }
#pragma unroll
    for (int off = 1; off < 16; off <<= 1) {
#pragma unroll
      for (int r = 0; r < 8; ++r) tsum[r] += __shfl_xor(tsum[r], off);
    }
#pragma unroll
    for (int r = 0; r < 8; ++r) l_i[r] = l_i[r] * alpha[r] + tsum[r];
#pragma unroll
    for (int nt = 0; nt < 8; ++nt) {
#pragma unroll
      for (int r = 0; r < 8; ++r) o[nt][r] *= alpha[r];
    }
    // P: C-layout -> LDS (padded rows, bank-conflict-free) -> A-fragment
#pragma unroll
    for (int r = 0; r < 8; ++r) {
      int row = r + 8 * half;
      Plds[row * PLDS_STRIDE_ + n]      = f2bf(p0v[r]);
      Plds[row * PLDS_STRIDE_ + 16 + n] = f2bf(p1v[r]);
    }
    v16bf pa = a_frag_bf(Plds + n * PLDS_STRIDE_, 0, half);
    // O += P * V   (V transposed: contiguous B-fragments)
#pragma unroll
    for (int nt = 0; nt < 8; ++nt) {
      const __bf16* vrow = Vt + (size_t)(win * D_ + nt * 16 + n) * LW_ + k0;
      o[nt] = wmma_bf16(pa, b_frag(vrow, 0, half), o[nt]);
    }
  }

  // normalize and scatter back to natural token order (inverse window map)
  float inv[8];
#pragma unroll
  for (int r = 0; r < 8; ++r) inv[r] = 1.0f / l_i[r];
#pragma unroll
  for (int r = 0; r < 8; ++r) {
    const int t = map_token(win, q0 + r + 8 * half, shift);
#pragma unroll
    for (int nt = 0; nt < 8; ++nt)
      Aout[(size_t)t * D_ + nt * 16 + n] = f2bf(o[nt][r] * inv[r]);
  }
}

// ---------------------------------------------------------------------------
// K3: message = LayerNorm(Aout @ Wm; g1,b1)  -> bf16 row-major
// ---------------------------------------------------------------------------
__global__ void __launch_bounds__(32)
k_msg_ln(const __bf16* __restrict__ Aout, const __bf16* __restrict__ Wmt,
         const float* __restrict__ g1, const float* __restrict__ b1,
         __bf16* __restrict__ Mout) {
  const int t0 = blockIdx.x << 4;
  const int lane = threadIdx.x;
  const int n = lane & 15, half = lane >> 4;

  const __bf16* arow = Aout + (size_t)(t0 + n) * D_;
  v16bf a[4];
#pragma unroll
  for (int kc = 0; kc < 4; ++kc) a[kc] = a_frag_bf(arow, kc * 32, half);

  v8f c[8];
  v8f z = {};
#pragma unroll
  for (int nt = 0; nt < 8; ++nt) {
    c[nt] = z;
    const __bf16* wrow = Wmt + (size_t)(nt * 16 + n) * D_;
#pragma unroll
    for (int kc = 0; kc < 4; ++kc)
      c[nt] = wmma_bf16(a[kc], b_frag(wrow, kc * 32, half), c[nt]);
  }
  // fused LayerNorm over D=128 (8 tiles x 16 lanes)
  float s1[8], s2[8];
#pragma unroll
  for (int r = 0; r < 8; ++r) { s1[r] = 0.f; s2[r] = 0.f; }
#pragma unroll
  for (int nt = 0; nt < 8; ++nt) {
#pragma unroll
    for (int r = 0; r < 8; ++r) { float x = c[nt][r]; s1[r] += x; s2[r] += x * x; }
  }
#pragma unroll
  for (int off = 1; off < 16; off <<= 1) {
#pragma unroll
    for (int r = 0; r < 8; ++r) {
      s1[r] += __shfl_xor(s1[r], off);
      s2[r] += __shfl_xor(s2[r], off);
    }
  }
  float mu[8], rstd[8];
#pragma unroll
  for (int r = 0; r < 8; ++r) {
    float mean = s1[r] * (1.0f / D_);
    float var = s2[r] * (1.0f / D_) - mean * mean;
    mu[r] = mean;
    rstd[r] = rsqrtf(fmaxf(var, 0.f) + LN_EPS_);
  }
  float gv[8], bv[8];
#pragma unroll
  for (int nt = 0; nt < 8; ++nt) { gv[nt] = g1[nt * 16 + n]; bv[nt] = b1[nt * 16 + n]; }
#pragma unroll
  for (int r = 0; r < 8; ++r) {
    const size_t t = (size_t)(t0 + r + 8 * half);
#pragma unroll
    for (int nt = 0; nt < 8; ++nt)
      Mout[t * D_ + nt * 16 + n] =
          f2bf((c[nt][r] - mu[r]) * rstd[r] * gv[nt] + bv[nt]);
  }
}

// ---------------------------------------------------------------------------
// K4: fused MLP + LayerNorm + residual.
//   out = src + LN(gelu([src||msg] @ W1) @ W2; g2,b2)
// Hidden activations (16x1024) never touch HBM: 32 chunks of 32 columns,
// each chunk GEMM -> gelu -> LDS relayout -> accumulate into output tiles.
// ---------------------------------------------------------------------------
__global__ void __launch_bounds__(32)
k_mlp(const float* __restrict__ src, const __bf16* __restrict__ Mout,
      const __bf16* __restrict__ W1t, const __bf16* __restrict__ W2t,
      const float* __restrict__ g2, const float* __restrict__ b2,
      float* __restrict__ out) {
  __shared__ __align__(16) __bf16 Plds[16 * PLDS_STRIDE_];
  const int t0 = blockIdx.x << 4;
  const int lane = threadIdx.x;
  const int n = lane & 15, half = lane >> 4;

  const float*  srow = src  + (size_t)(t0 + n) * D_;
  const __bf16* mrow = Mout + (size_t)(t0 + n) * D_;
  v16bf a1[4], a2[4];
#pragma unroll
  for (int kc = 0; kc < 4; ++kc) {
    a1[kc] = a_frag_f32(srow, kc * 32, half);   // features 0..127 (source)
    a2[kc] = a_frag_bf(mrow, kc * 32, half);    // features 128..255 (message)
  }

  v8f oc[8];
  v8f z = {};
#pragma unroll
  for (int nt = 0; nt < 8; ++nt) oc[nt] = z;

  for (int hc = 0; hc < 32; ++hc) {
#pragma unroll
    for (int st = 0; st < 2; ++st) {
      const __bf16* w1row = W1t + (size_t)((hc * 2 + st) * 16 + n) * 256;
      v8f h = z;
#pragma unroll
      for (int kc = 0; kc < 4; ++kc)
        h = wmma_bf16(a1[kc], b_frag(w1row, kc * 32, half), h);
#pragma unroll
      for (int kc = 0; kc < 4; ++kc)
        h = wmma_bf16(a2[kc], b_frag(w1row, 128 + kc * 32, half), h);
      // exact gelu, store C-layout tile into LDS
#pragma unroll
      for (int r = 0; r < 8; ++r) {
        float x = h[r];
        float g = 0.5f * x * (1.0f + erff(x * 0.70710678118654752f));
        Plds[(r + 8 * half) * PLDS_STRIDE_ + st * 16 + n] = f2bf(g);
      }
    }
    v16bf pa = a_frag_bf(Plds + n * PLDS_STRIDE_, 0, half);
#pragma unroll
    for (int nt = 0; nt < 8; ++nt) {
      const __bf16* w2row = W2t + (size_t)(nt * 16 + n) * HID_ + hc * 32;
      oc[nt] = wmma_bf16(pa, b_frag(w2row, 0, half), oc[nt]);
    }
  }

  // fused LayerNorm + residual add, fp32 output
  float s1[8], s2[8];
#pragma unroll
  for (int r = 0; r < 8; ++r) { s1[r] = 0.f; s2[r] = 0.f; }
#pragma unroll
  for (int nt = 0; nt < 8; ++nt) {
#pragma unroll
    for (int r = 0; r < 8; ++r) { float x = oc[nt][r]; s1[r] += x; s2[r] += x * x; }
  }
#pragma unroll
  for (int off = 1; off < 16; off <<= 1) {
#pragma unroll
    for (int r = 0; r < 8; ++r) {
      s1[r] += __shfl_xor(s1[r], off);
      s2[r] += __shfl_xor(s2[r], off);
    }
  }
  float mu[8], rstd[8];
#pragma unroll
  for (int r = 0; r < 8; ++r) {
    float mean = s1[r] * (1.0f / D_);
    float var = s2[r] * (1.0f / D_) - mean * mean;
    mu[r] = mean;
    rstd[r] = rsqrtf(fmaxf(var, 0.f) + LN_EPS_);
  }
  float gv[8], bv[8];
#pragma unroll
  for (int nt = 0; nt < 8; ++nt) { gv[nt] = g2[nt * 16 + n]; bv[nt] = b2[nt * 16 + n]; }
#pragma unroll
  for (int r = 0; r < 8; ++r) {
    const size_t t = (size_t)(t0 + r + 8 * half);
#pragma unroll
    for (int nt = 0; nt < 8; ++nt) {
      const int d = nt * 16 + n;
      float y = (oc[nt][r] - mu[r]) * rstd[r] * gv[nt] + bv[nt];
      out[t * D_ + d] = src[t * D_ + d] + y;
    }
  }
}

// ---------------------------------------------------------------------------
// launch
// ---------------------------------------------------------------------------
extern "C" void kernel_launch(void* const* d_in, const int* in_sizes, int n_in,
                              void* d_out, int out_size, void* d_ws, size_t ws_size,
                              hipStream_t stream) {
  (void)in_sizes; (void)n_in; (void)out_size; (void)ws_size;
  const float* src  = (const float*)d_in[0];
  // d_in[1] target == source (self-attention branch); d_in[3..6] unused consts
  const float* mask = (const float*)d_in[2];
  const int* p_shift = (const int*)d_in[7];
  const float* Wq = (const float*)d_in[8];
  const float* Wk = (const float*)d_in[9];
  const float* Wv = (const float*)d_in[10];
  const float* Wm = (const float*)d_in[11];
  const float* g1 = (const float*)d_in[12];
  const float* b1 = (const float*)d_in[13];
  const float* W1 = (const float*)d_in[14];
  const float* W2 = (const float*)d_in[15];
  const float* g2 = (const float*)d_in[16];
  const float* b2 = (const float*)d_in[17];
  float* out = (float*)d_out;

  char* ws = (char*)d_ws;
  __bf16* Wqt  = (__bf16*)(ws + 0);
  __bf16* Wkt  = (__bf16*)(ws + 32768);
  __bf16* Wvt  = (__bf16*)(ws + 65536);
  __bf16* Wmt  = (__bf16*)(ws + 98304);
  __bf16* W1t  = (__bf16*)(ws + 131072);
  __bf16* W2t  = (__bf16*)(ws + 655360);
  size_t base = 917504;                       // 896 KB of weights above
  __bf16* Qw   = (__bf16*)(ws + base + 0ull * 8388608ull);
  __bf16* Kw   = (__bf16*)(ws + base + 1ull * 8388608ull);
  __bf16* Vt   = (__bf16*)(ws + base + 2ull * 8388608ull);
  __bf16* Aout = (__bf16*)(ws + base + 3ull * 8388608ull);
  __bf16* Mout = (__bf16*)(ws + base + 4ull * 8388608ull);

  k_prep_weights<<<1792, 256, 0, stream>>>(Wq, Wk, Wv, Wm, W1, W2,
                                           Wqt, Wkt, Wvt, Wmt, W1t, W2t);
  k_qkv_proj<<<6144, 32, 0, stream>>>(src, Wqt, Wkt, Wvt, Qw, Kw, Vt, p_shift);
  k_attn<<<2048, 32, 0, stream>>>(Qw, Kw, Vt, mask, Aout, p_shift);
  k_msg_ln<<<2048, 32, 0, stream>>>(Aout, Wmt, g1, b1, Mout);
  k_mlp<<<2048, 32, 0, stream>>>(src, Mout, W1t, W2t, g2, b2, out);
}
